// AttentionModule_80633716015204
// MI455X (gfx1250) — matrix operand order repaired and checked by reading, not compiled, not run on hardware
//
#include <hip/hip_runtime.h>
#include <hip/hip_bf16.h>

// ---------------------------------------------------------------------------
// Attention module for MI455X (gfx1250, wave32, WMMA).
// Dominant cost: e_proj GEMM [32768,1024]x[1024,2048] = 137 GFLOP, fused with
// tanh + w2 reduction so the 268 MB "act" tensor is never written.
// bf16 WMMA (f32 accumulate). All static operands (Wa slice, Wc, combined)
// are pre-converted to bf16 once, so both WMMA hot loops are pure
// {2x ds/global b128 A-loads, 2x global b128 B-loads, v_wmma}.
// encoder_states (134 MB) fits in the 192 MB L2 -> second pass reads L2.
// ---------------------------------------------------------------------------

#define B_  64
#define T_  512
#define E_  1024
#define D_  1024
#define F2_ 2048   // 2E
#define DE_ 2048   // D+E

typedef __attribute__((ext_vector_type(16))) __bf16 v16bf;
typedef __attribute__((ext_vector_type(8)))  float  v8f;

union BF16x16 {
    v16bf          v;
    unsigned short u[16];
    uint4          q[2];
};
static_assert(sizeof(BF16x16) == 32, "v16bf must be 32 bytes");

__device__ __forceinline__ unsigned short f2bf_bits(float f) {
    unsigned x = __builtin_bit_cast(unsigned, f);
    x = x + 0x7FFFu + ((x >> 16) & 1u);   // round-to-nearest-even
    return (unsigned short)(x >> 16);
}

__device__ __forceinline__ void pack4(unsigned short* u, float4 c) {
    u[0] = f2bf_bits(c.x); u[1] = f2bf_bits(c.y);
    u[2] = f2bf_bits(c.z); u[3] = f2bf_bits(c.w);
}
__device__ __forceinline__ void pack8(unsigned short* u, const float* p) {
    pack4(u,     *(const float4*)(p));
    pack4(u + 4, *(const float4*)(p + 4));
}
__device__ __forceinline__ void pack16(unsigned short* u, const float* p) {
    pack8(u, p); pack8(u + 8, p + 8);
}

// Branch-free tanh: t = exp(-2|x|) in (0,1] (no overflow path), then
// tanh(|x|) = (1-t)/(1+t); restore sign. ~7 VALU ops, one v_exp_f32.
__device__ __forceinline__ float tanh_fast(float x) {
    float ax = __builtin_fabsf(x);
    float t  = __expf(-2.0f * ax);
    float r  = __fdividef(1.0f - t, 1.0f + t);
    return copysignf(r, x);
}

// ---------------------------------------------------------------------------
// Kernel 0a: one-shot f32 -> bf16 of Wa[:, D:]  (2048 x 1024, 4 MB out).
// ---------------------------------------------------------------------------
__global__ __launch_bounds__(256) void wa_bf16_kernel(
    const float* __restrict__ Wa, unsigned short* __restrict__ WaBf)
{
    int idx = blockIdx.x * 256 + threadIdx.x;        // 2048*1024
    int f = idx >> 10;
    int k = idx & (E_ - 1);
    WaBf[idx] = f2bf_bits(Wa[(size_t)f * DE_ + D_ + k]);
}

// Kernel 0b: one-shot f32 -> bf16 flat convert of Wc (1024 x 2048).
__global__ __launch_bounds__(256) void wc_bf16_kernel(
    const float* __restrict__ Wc, unsigned short* __restrict__ WcBf)
{
    int idx = blockIdx.x * 256 + threadIdx.x;        // 1024*2048
    WcBf[idx] = f2bf_bits(Wc[idx]);
}

// ---------------------------------------------------------------------------
// Kernel 1: h_proj[b,f] = hidden[b,:] . Wa[f,0:D] + ba[f]      (64 x 2048)
// ---------------------------------------------------------------------------
__global__ __launch_bounds__(256) void hproj_kernel(
    const float* __restrict__ hidden, const float* __restrict__ Wa,
    const float* __restrict__ ba, float* __restrict__ hproj)
{
    int idx = blockIdx.x * 256 + threadIdx.x;      // 64*2048 = 131072
    int b = idx >> 11;
    int f = idx & (F2_ - 1);
    const float* hr = hidden + (size_t)b * D_;
    const float* wr = Wa + (size_t)f * DE_;        // columns 0..D-1
    float acc = ba[f];
    for (int k = 0; k < D_; k += 4) {
        float4 h = *(const float4*)(hr + k);
        float4 w = *(const float4*)(wr + k);
        acc += h.x * w.x + h.y * w.y + h.z * w.z + h.w * w.w;
    }
    hproj[idx] = acc;
}

// ---------------------------------------------------------------------------
// Kernel 2 (hot): fused  scores[b,t] = sum_f tanh(hproj[b,f] +
//                        (enc[t,b,:] . Wa[f,D:].T)[f]) * w2[f] + b2
// One workgroup per 16-row M-tile of A = enc viewed as [T*B, E].
// A tile staged once in LDS as bf16 (pitch 1032 -> conflict-free b128 reads).
// 8 waves x 16 N-tiles x 32 K-steps of v_wmma_f32_16x16x32_bf16.
// PRE=true: B fragments are direct b128 loads from pre-converted bf16 Wa.
// ---------------------------------------------------------------------------
#define APITCH 1032   // halves; 2064 B/row -> 4-bank shift per row

template <bool PRE>
__global__ __launch_bounds__(256) void attn_scores_kernel(
    const float* __restrict__ enc, const float* __restrict__ Wa,
    const unsigned short* __restrict__ WaBf,
    const float* __restrict__ hproj, const float* __restrict__ w2,
    const float* __restrict__ b2, float* __restrict__ scores)
{
    __shared__ unsigned short sA[16 * APITCH];
    __shared__ float sScore[16];

    const int tid   = threadIdx.x;
    const int wave  = tid >> 5;
    const int lane  = tid & 31;
    const int sub   = lane >> 4;       // half-wave: 0 or 1
    const int ln    = lane & 15;
    const int m0    = blockIdx.x * 16; // 16 aligned rows share the same t
    const int tIdx  = m0 / B_;
    const int bBase = m0 % B_;

    // Stage A tile (16 x 1024) -> bf16 LDS, coalesced over K.
    for (int idx = tid; idx < 16 * E_; idx += 256) {
        int r = idx >> 10;
        int k = idx & (E_ - 1);
        sA[r * APITCH + k] = f2bf_bits(enc[(size_t)(m0 + r) * E_ + k]);
    }
    if (tid < 16) sScore[tid] = 0.0f;
    __syncthreads();

    float rowAcc[8];
#pragma unroll
    for (int j = 0; j < 8; ++j) rowAcc[j] = 0.0f;

    // Each wave owns 256 f-columns = 16 N-tiles.
    for (int nt = 0; nt < 16; ++nt) {
        const int f = wave * 256 + nt * 16 + ln;       // this lane's column
        const float*          brow  = Wa + (size_t)f * DE_ + D_;  // Wa[f, D:]
        const unsigned short* browb = WaBf + (size_t)f * E_;
        v8f acc = {};
#pragma unroll 4
        for (int k0 = 0; k0 < E_; k0 += 32) {
            // A fragment: lane holds 16 consecutive K (matches B).
            BF16x16 a;
            const unsigned short* ap = &sA[ln * APITCH + k0 + 16 * sub];
            a.q[0] = *(const uint4*)(ap);
            a.q[1] = *(const uint4*)(ap + 8);
            // B fragment: lane holds 16 consecutive K for fixed column f.
            BF16x16 bb;
            if (PRE) {
                const uint4* bq = (const uint4*)(browb + k0 + 16 * sub);
                bb.q[0] = bq[0];
                bb.q[1] = bq[1];
            } else {
                pack16(bb.u, brow + k0 + 16 * sub);
            }
            acc = __builtin_amdgcn_wmma_f32_16x16x32_bf16(
                false, a.v, false, bb.v, (short)0, acc, false, false);
        }
        // Epilogue: tanh + w2 weighting, accumulate per output row.
        const float w2f = w2[f];
#pragma unroll
        for (int j = 0; j < 8; ++j) {
            int b = bBase + j + 8 * sub;               // C row = j + 8*sub
            rowAcc[j] += tanh_fast(acc[j] + hproj[(size_t)b * F2_ + f]) * w2f;
        }
    }

    // Reduce over the 16 columns held by each half-wave.
#pragma unroll
    for (int j = 0; j < 8; ++j) {
        float v = rowAcc[j];
        v += __shfl_xor(v, 1, 32);
        v += __shfl_xor(v, 2, 32);
        v += __shfl_xor(v, 4, 32);
        v += __shfl_xor(v, 8, 32);
        rowAcc[j] = v;
    }
    if (ln == 0) {
#pragma unroll
        for (int j = 0; j < 8; ++j)
            atomicAdd(&sScore[j + 8 * sub], rowAcc[j]);   // ds_add_f32
    }
    __syncthreads();
    if (tid < 16) {
        int b = bBase + tid;
        scores[(size_t)b * T_ + tIdx] = sScore[tid] + b2[0];
    }
}

// ---------------------------------------------------------------------------
// Kernel 3: in-place softmax over T per batch row. One block per row.
// ---------------------------------------------------------------------------
__global__ __launch_bounds__(256) void softmax_kernel(float* __restrict__ scores)
{
    __shared__ float red[256];
    float* row = scores + (size_t)blockIdx.x * T_;
    int tid = threadIdx.x;
    float a = row[tid], b = row[tid + 256];
    red[tid] = fmaxf(a, b);
    __syncthreads();
    for (int s = 128; s > 0; s >>= 1) {
        if (tid < s) red[tid] = fmaxf(red[tid], red[tid + s]);
        __syncthreads();
    }
    float mx = red[0];
    __syncthreads();
    float ea = __expf(a - mx), eb = __expf(b - mx);
    red[tid] = ea + eb;
    __syncthreads();
    for (int s = 128; s > 0; s >>= 1) {
        if (tid < s) red[tid] += red[tid + s];
        __syncthreads();
    }
    float inv = 1.0f / red[0];
    row[tid]       = ea * inv;
    row[tid + 256] = eb * inv;
}

// ---------------------------------------------------------------------------
// Kernel 4: applied[b,e] = sum_t w[b,t] * enc[t,b,e]; also assembles
// combined = [decoder_out | applied] (f32, and bf16 copy when PRE).
// enc re-read hits the 192 MB L2.
// ---------------------------------------------------------------------------
template <bool PRE>
__global__ __launch_bounds__(256) void applied_kernel(
    const float* __restrict__ weights, const float* __restrict__ enc,
    const float* __restrict__ decoder_out,
    float* __restrict__ applied_out, float* __restrict__ combined,
    unsigned short* __restrict__ combinedBf)
{
    __shared__ float w[T_];
    const int b = blockIdx.x >> 2;
    const int e = ((blockIdx.x & 3) << 8) + threadIdx.x;
    w[threadIdx.x]       = weights[(size_t)b * T_ + threadIdx.x];
    w[threadIdx.x + 256] = weights[(size_t)b * T_ + 256 + threadIdx.x];
    __syncthreads();
    float acc = 0.0f;
    for (int t = 0; t < T_; ++t)
        acc += w[t] * enc[((size_t)t * B_ + b) * E_ + e];
    const float dec = decoder_out[(size_t)b * D_ + e];
    applied_out[(size_t)b * E_ + e]    = acc;                  // output #2
    combined[(size_t)b * DE_ + D_ + e] = acc;
    combined[(size_t)b * DE_ + e]      = dec;
    if (PRE) {
        combinedBf[(size_t)b * DE_ + D_ + e] = f2bf_bits(acc);
        combinedBf[(size_t)b * DE_ + e]      = f2bf_bits(dec);
    }
}

// ---------------------------------------------------------------------------
// Kernel 5: out = tanh(combined @ Wc.T + bc), 64x1024x2048 via bf16 WMMA.
// PRE=true: both fragments are direct b128 loads from bf16 buffers
// (combinedBf 256 KB + WcBf 4 MB, both L2-resident).
// ---------------------------------------------------------------------------
template <bool PRE>
__global__ __launch_bounds__(256) void out_gemm_kernel(
    const float* __restrict__ combined, const unsigned short* __restrict__ combinedBf,
    const float* __restrict__ Wc, const unsigned short* __restrict__ WcBf,
    const float* __restrict__ bc, float* __restrict__ out)
{
    const int tid  = threadIdx.x;
    const int wave = tid >> 5;
    const int lane = tid & 31;
    const int sub  = lane >> 4;
    const int ln   = lane & 15;
    const int m0   = blockIdx.x * 16;                 // 4 blocks cover M=64

    const float*          arow  = combined   + (size_t)(m0 + ln) * DE_;
    const unsigned short* arowb = combinedBf + (size_t)(m0 + ln) * DE_;
    for (int nt = 0; nt < 8; ++nt) {                  // 8 N-tiles per wave
        const int n = wave * 128 + nt * 16 + ln;
        const float*          brow  = Wc   + (size_t)n * DE_;
        const unsigned short* browb = WcBf + (size_t)n * DE_;
        v8f acc = {};
#pragma unroll 4
        for (int k0 = 0; k0 < DE_; k0 += 32) {
            BF16x16 a, bb;
            if (PRE) {
                a.q[0]  = *(const uint4*)(arowb + k0 + 16 * sub);
                a.q[1]  = *(const uint4*)(arowb + k0 + 16 * sub + 8);
                const uint4* bq = (const uint4*)(browb + k0 + 16 * sub);
                bb.q[0] = bq[0];
                bb.q[1] = bq[1];
            } else {
                pack8(a.u,     arow + k0 + 16 * sub);  // 16 consecutive K
                pack8(a.u + 8, arow + k0 + 16 * sub + 8);
                pack16(bb.u,   brow + k0 + 16 * sub);
            }
            acc = __builtin_amdgcn_wmma_f32_16x16x32_bf16(
                false, a.v, false, bb.v, (short)0, acc, false, false);
        }
        const float bias = bc[n];
#pragma unroll
        for (int j = 0; j < 8; ++j) {
            int row = m0 + j + 8 * sub;
            out[(size_t)row * D_ + n] = tanh_fast(acc[j] + bias);
        }
    }
}

// ---------------------------------------------------------------------------
extern "C" void kernel_launch(void* const* d_in, const int* in_sizes, int n_in,
                              void* d_out, int out_size, void* d_ws, size_t ws_size,
                              hipStream_t stream) {
    (void)in_sizes; (void)n_in; (void)out_size;
    const float* hidden      = (const float*)d_in[0];
    const float* decoder_out = (const float*)d_in[1];
    const float* enc         = (const float*)d_in[2];
    const float* Wa          = (const float*)d_in[3];
    const float* ba          = (const float*)d_in[4];
    const float* w2          = (const float*)d_in[5];
    const float* b2          = (const float*)d_in[6];
    const float* Wc          = (const float*)d_in[7];
    const float* bc          = (const float*)d_in[8];

    float* out         = (float*)d_out;              // [64,1024]
    float* applied_out = out + (size_t)B_ * D_;      // [64,1024]

    float* ws       = (float*)d_ws;
    float* hproj    = ws;                            // 131072 f32
    float* scores   = ws + 131072;                   //  32768 f32 (-> weights)
    float* combined = ws + 131072 + 32768;           // 131072 f32
    unsigned short* ush = (unsigned short*)(ws + 294912);
    unsigned short* WaBf       = ush;                          // 2M halves
    unsigned short* combinedBf = ush + (size_t)F2_ * E_;       // 128K halves
    unsigned short* WcBf       = combinedBf + (size_t)B_ * DE_;// 2M halves

    const size_t needed = 294912u * sizeof(float)
                        + ((size_t)F2_ * E_ + (size_t)B_ * DE_ + (size_t)D_ * DE_)
                          * sizeof(unsigned short);
    const bool pre = ws_size >= needed;

    hproj_kernel<<<512, 256, 0, stream>>>(hidden, Wa, ba, hproj);
    if (pre) {
        wa_bf16_kernel<<<(F2_ * E_) / 256, 256, 0, stream>>>(Wa, WaBf);
        wc_bf16_kernel<<<(D_ * DE_) / 256, 256, 0, stream>>>(Wc, WcBf);
        attn_scores_kernel<true><<<2048, 256, 0, stream>>>(
            enc, Wa, WaBf, hproj, w2, b2, scores);
        softmax_kernel<<<B_, 256, 0, stream>>>(scores);
        applied_kernel<true><<<B_ * 4, 256, 0, stream>>>(
            scores, enc, decoder_out, applied_out, combined, combinedBf);
        out_gemm_kernel<true><<<4, 256, 0, stream>>>(
            combined, combinedBf, Wc, WcBf, bc, out);
    } else {
        attn_scores_kernel<false><<<2048, 256, 0, stream>>>(
            enc, Wa, WaBf, hproj, w2, b2, scores);
        softmax_kernel<<<B_, 256, 0, stream>>>(scores);
        applied_kernel<false><<<B_ * 4, 256, 0, stream>>>(
            scores, enc, decoder_out, applied_out, combined, combinedBf);
        out_gemm_kernel<false><<<4, 256, 0, stream>>>(
            combined, combinedBf, Wc, WcBf, bc, out);
    }
}
